// GAT_56246891708452
// MI455X (gfx1250) — matrix operand order repaired and checked by reading, not compiled, not run on hardware
//
#include <hip/hip_runtime.h>
#include <math.h>
#include <stdint.h>

// ---------------------------------------------------------------------------
// Types / constants
// ---------------------------------------------------------------------------
typedef _Float16 half_t;
typedef __attribute__((ext_vector_type(16))) _Float16 v16h;
typedef __attribute__((ext_vector_type(8)))  _Float16 v8h;
typedef __attribute__((ext_vector_type(8)))  float    v8f;
typedef __attribute__((ext_vector_type(4)))  unsigned int u32x4;
typedef __attribute__((ext_vector_type(8)))  int      i32x8;
typedef __attribute__((ext_vector_type(4)))  int      i32x4;

#define NODES  20000
#define EDGES  160000
#define NGRAPH 200
#define GPN    100

static constexpr float kBNInv = 0.9999950000374996f;  // 1/sqrt(1+1e-5)

static inline int cdiv(int a, int b) { return (a + b - 1) / b; }

#if defined(__has_builtin)
#if __has_builtin(__builtin_amdgcn_tensor_load_to_lds) && \
    __has_builtin(__builtin_amdgcn_s_wait_tensorcnt)
#define USE_TDM 1
#endif
#endif

// ---------------------------------------------------------------------------
// Weight staging: TDM (Tensor Data Mover) async copy of the 64xCIN f16 weight
// tile from global to LDS, issued once by wave 0; the workgroup barrier after
// staging publishes it. Falls back to a cooperative vector copy if the TDM
// builtin is unavailable. (This toolchain uses the 6-argument builtin:
// g0, g1, g2, g3, g4, cpol.)
// ---------------------------------------------------------------------------
template <int CIN>
__device__ inline void stage_w(half_t* sW, const half_t* __restrict__ W) {
#ifdef USE_TDM
  if (threadIdx.x < 32) {
    const unsigned long long ga = (unsigned long long)(uintptr_t)W;
    const unsigned lds = (unsigned)(uintptr_t)sW;  // byte offset in LDS space
    u32x4 g0;
    g0[0] = 1u;                                    // count=1 (valid descriptor)
    g0[1] = lds;                                   // lds_addr
    g0[2] = (unsigned)(ga & 0xffffffffu);          // global_addr[31:0]
    g0[3] = (unsigned)((ga >> 32) & 0x01ffffffu)   // global_addr[56:32]
            | 0x80000000u;                         // type=2 ("image")
    i32x8 g1;
    g1[0] = 0x00010000;                            // data_size=1 -> 2 bytes
    g1[1] = (int)((unsigned)CIN << 16);            // tensor_dim0 = CIN
    g1[2] = (int)(64u << 16);                      // tensor_dim1 = 64
    g1[3] = (int)((unsigned)CIN << 16);            // tile_dim0 = CIN
    g1[4] = 64;                                    // tile_dim1 = 64
    g1[5] = CIN;                                   // tensor_dim0_stride = CIN
    g1[6] = 0;
    g1[7] = 0;
    const i32x4 z4 = {0, 0, 0, 0};                 // groups 2/3 unused (2-D)
    const i32x8 z8 = {0, 0, 0, 0, 0, 0, 0, 0};
    __builtin_amdgcn_tensor_load_to_lds(g0, g1, z4, z4, z8, 0);
    __builtin_amdgcn_s_wait_tensorcnt(0);
  }
#else
  for (int i = threadIdx.x; i < 64 * CIN / 8; i += 256)
    *(v8h*)(sW + i * 8) = *(const v8h*)(W + i * 8);
#endif
}

// ---------------------------------------------------------------------------
// WMMA core: each block (256 thr = 8 waves) computes a 32x64 f16 output tile.
// wave w: row-tile rt = w>>2 (0..1), col-tile ct = w&3 (0..3).
// sX: 32 x CIN f16 (row major), sW: 64 x CIN f16 (row major, B = W^T).
// Epilogue applies y = (acc + b) * (g*BNinv) + be, optional ELU, stores f16.
// ---------------------------------------------------------------------------
template <int CIN, bool ELU>
__device__ inline void gemm_core(const half_t* sX, const half_t* sW,
                                 const float* __restrict__ bias,
                                 const float* __restrict__ gain,
                                 const float* __restrict__ beta,
                                 half_t* __restrict__ Y, int row0, int M) {
  const int tid  = threadIdx.x;
  const int wave = tid >> 5;
  const int lane = tid & 31;
  const int rt = wave >> 2;
  const int ct = wave & 3;
  const int r = lane & 15;
  const int h = lane >> 4;

  v8f acc = {};
#pragma unroll
  for (int k0 = 0; k0 < CIN; k0 += 32) {
    union { v16h v; v8h h8[2]; } a, b;
    // A fragment (16x32 f16): lane half h holds K = k0+h*16..+15
    const half_t* ap = sX + (rt * 16 + r) * CIN + k0;
    a.v = *(const v16h*)(ap + h * 16);
    // B fragment (32x16 f16): lane holds 16 contiguous K of output column n
    b.v = *(const v16h*)(sW + (ct * 16 + r) * CIN + k0 + h * 16);
    acc = __builtin_amdgcn_wmma_f32_16x16x32_f16(false, a.v, false, b.v,
                                                 (short)0, acc, false, false);
  }

  // D layout: lane holds column n = ct*16 + (lane&15); rows m = h*8 + 0..7
  const int n = ct * 16 + r;
  const float bb = bias[n];
  const float gg = gain[n] * kBNInv;
  const float bt = beta[n];
#pragma unroll
  for (int vv = 0; vv < 8; ++vv) {
    const int m = row0 + rt * 16 + h * 8 + vv;
    if (m < M) {
      float y = (acc[vv] + bb) * gg + bt;
      if (ELU) y = (y > 0.f) ? y : (__expf(y) - 1.f);
      Y[(size_t)m * 64 + n] = (half_t)y;
    }
  }
}

// ---------------------------------------------------------------------------
// Generic GEMM: X is a dense M x CIN f16 matrix.
// ---------------------------------------------------------------------------
template <int CIN, bool ELU>
__global__ __launch_bounds__(256) void k_gemm(const half_t* __restrict__ X,
                                              const half_t* __restrict__ W,
                                              const float* __restrict__ b,
                                              const float* __restrict__ g,
                                              const float* __restrict__ be,
                                              half_t* __restrict__ Y, int M) {
  __shared__ __align__(32) half_t sX[32 * CIN];
  __shared__ __align__(32) half_t sW[64 * CIN];
  const int row0 = blockIdx.x * 32;
  stage_w<CIN>(sW, W);
  constexpr int CH = CIN / 8;
  for (int i = threadIdx.x; i < 32 * CH; i += 256) {
    const int r = i / CH, c = (i % CH) * 8;
    v8h v = {};
    if (row0 + r < M) v = *(const v8h*)(X + (size_t)(row0 + r) * CIN + c);
    *(v8h*)(sX + r * CIN + c) = v;
  }
  __syncthreads();
  gemm_core<CIN, ELU>(sX, sW, b, g, be, Y, row0, M);
}

// ---------------------------------------------------------------------------
// edge_model block0: fused gather of [x[dst] | x[src] | ea*dij | u[batch[src]]]
// (CIN = 256, ELU).
// ---------------------------------------------------------------------------
__global__ __launch_bounds__(256) void k_edge_gemm(
    const half_t* __restrict__ xh, const half_t* __restrict__ eah,
    const half_t* __restrict__ uh, const int* __restrict__ ei,
    const int* __restrict__ batch, const float* __restrict__ dij,
    const half_t* __restrict__ W, const float* __restrict__ b,
    const float* __restrict__ g, const float* __restrict__ be,
    half_t* __restrict__ Y, int E) {
  constexpr int CIN = 256;
  __shared__ __align__(32) half_t sX[32 * CIN];
  __shared__ __align__(32) half_t sW[64 * CIN];
  const int row0 = blockIdx.x * 32;
  stage_w<CIN>(sW, W);
  for (int i = threadIdx.x; i < 32 * 32; i += 256) {
    const int r = i >> 5, c = i & 31;  // chunk of 8 halves
    const int e = row0 + r;
    v8h v = {};
    if (e < E) {
      const int seg = c >> 3, o = (c & 7) * 8;
      if (seg == 0) {
        v = *(const v8h*)(xh + (size_t)ei[E + e] * 64 + o);        // x[dst]
      } else if (seg == 1) {
        v = *(const v8h*)(xh + (size_t)ei[e] * 64 + o);            // x[src]
      } else if (seg == 2) {
        v = *(const v8h*)(eah + (size_t)e * 64 + o);               // ea*dij
        const float d = dij[e];
#pragma unroll
        for (int q = 0; q < 8; ++q) v[q] = (half_t)((float)v[q] * d);
      } else {
        v = *(const v8h*)(uh + (size_t)batch[ei[e]] * 64 + o);     // u[batch[src]]
      }
    }
    *(v8h*)(sX + r * CIN + c * 8) = v;
  }
  __syncthreads();
  gemm_core<CIN, true>(sX, sW, b, g, be, Y, row0, E);
}

// ---------------------------------------------------------------------------
// Concat-of-two-64 GEMM: [A | B] -> CIN = 128.
// ---------------------------------------------------------------------------
template <bool ELU>
__global__ __launch_bounds__(256) void k_cat_gemm(
    const half_t* __restrict__ A, const half_t* __restrict__ Bm,
    const half_t* __restrict__ W, const float* __restrict__ b,
    const float* __restrict__ g, const float* __restrict__ be,
    half_t* __restrict__ Y, int M) {
  constexpr int CIN = 128;
  __shared__ __align__(32) half_t sX[32 * CIN];
  __shared__ __align__(32) half_t sW[64 * CIN];
  const int row0 = blockIdx.x * 32;
  stage_w<CIN>(sW, W);
  for (int i = threadIdx.x; i < 32 * 16; i += 256) {
    const int r = i >> 4, c = i & 15;
    const int m = row0 + r;
    v8h v = {};
    if (m < M) {
      const half_t* src = (c < 8) ? (A + (size_t)m * 64 + c * 8)
                                  : (Bm + (size_t)m * 64 + (c - 8) * 8);
      v = *(const v8h*)src;
    }
    *(v8h*)(sX + r * CIN + c * 8) = v;
  }
  __syncthreads();
  gemm_core<CIN, ELU>(sX, sW, b, g, be, Y, row0, M);
}

// ---------------------------------------------------------------------------
// node_model block0: fused gather of [x | u[batch] | agg(f32)] -> CIN = 192.
// ---------------------------------------------------------------------------
__global__ __launch_bounds__(256) void k_node_gemm(
    const half_t* __restrict__ xh, const half_t* __restrict__ uh,
    const float* __restrict__ agg, const int* __restrict__ batch,
    const half_t* __restrict__ W, const float* __restrict__ b,
    const float* __restrict__ g, const float* __restrict__ be,
    half_t* __restrict__ Y, int Nn) {
  constexpr int CIN = 192;
  __shared__ __align__(32) half_t sX[32 * CIN];
  __shared__ __align__(32) half_t sW[64 * CIN];
  const int row0 = blockIdx.x * 32;
  stage_w<CIN>(sW, W);
  for (int i = threadIdx.x; i < 32 * 24; i += 256) {
    const int r = i / 24, c = i % 24;
    const int m = row0 + r;
    v8h v = {};
    if (m < Nn) {
      if (c < 8) {
        v = *(const v8h*)(xh + (size_t)m * 64 + c * 8);
      } else if (c < 16) {
        v = *(const v8h*)(uh + (size_t)batch[m] * 64 + (c - 8) * 8);
      } else {
        const float* ap = agg + (size_t)m * 64 + (c - 16) * 8;
#pragma unroll
        for (int q = 0; q < 8; ++q) v[q] = (half_t)ap[q];
      }
    }
    *(v8h*)(sX + r * CIN + c * 8) = v;
  }
  __syncthreads();
  gemm_core<CIN, true>(sX, sW, b, g, be, Y, row0, Nn);
}

// ---------------------------------------------------------------------------
// Elementwise / scatter helpers
// ---------------------------------------------------------------------------
__global__ void k_f2h(const float* __restrict__ s, half_t* __restrict__ d, int n) {
  int i = blockIdx.x * 256 + threadIdx.x;
  if (i < n) d[i] = (half_t)s[i];
}

__global__ void k_zero(float* __restrict__ p, int n) {
  int i = blockIdx.x * 256 + threadIdx.x;
  if (i < n) p[i] = 0.f;
}

// agg[src[e]] += ea_new[e] * dij[e]
__global__ void k_edge_agg(const half_t* __restrict__ ea,
                           const float* __restrict__ dij,
                           const int* __restrict__ ei,
                           float* __restrict__ agg, int E) {
  int idx = blockIdx.x * 256 + threadIdx.x;
  if (idx >= E * 64) return;
  const int e = idx >> 6, f = idx & 63;
  const float v = (float)ea[idx] * dij[e];
  __hip_atomic_fetch_add(&agg[(size_t)ei[e] * 64 + f], v, __ATOMIC_RELAXED,
                         __HIP_MEMORY_SCOPE_AGENT);
}

// raw attention + pij logits (post leaky-relu)
__global__ void k_att_raw(const half_t* __restrict__ xn, const int* __restrict__ gi,
                          const float* __restrict__ attg, const float* __restrict__ pijw,
                          const float* __restrict__ Pij, float* __restrict__ att,
                          float* __restrict__ pw, int Ne) {
  int e = blockIdx.x * 256 + threadIdx.x;
  if (e >= Ne) return;
  const int a0 = gi[e], a1 = gi[Ne + e];
  float s = 0.f;
  for (int c = 0; c < 64; ++c)
    s += (float)xn[(size_t)a0 * 64 + c] * attg[c] +
         (float)xn[(size_t)a1 * 64 + c] * attg[64 + c];
  att[e] = (s > 0.f) ? s : 0.2f * s;
  float p = Pij[e * 3] * pijw[0] + Pij[e * 3 + 1] * pijw[1] + Pij[e * 3 + 2] * pijw[2];
  pw[e] = (p > 0.f) ? p : 0.2f * p;
}

// per-graph (100 contiguous edges) double softmax + combine: one block/graph
__global__ void k_softmax_w(const float* __restrict__ att, const float* __restrict__ pw,
                            const float* __restrict__ wij, float* __restrict__ w) {
  __shared__ float red[128];
  const int b = blockIdx.x, t = threadIdx.x;
  const int e = b * GPN + t;
  const float a = (t < GPN) ? att[e] : -1e30f;
  const float p = (t < GPN) ? pw[e] : -1e30f;

  red[t] = a; __syncthreads();
  for (int s = 64; s > 0; s >>= 1) { if (t < s) red[t] = fmaxf(red[t], red[t + s]); __syncthreads(); }
  const float amax = red[0]; __syncthreads();
  const float ea_ = (t < GPN) ? __expf(a - amax) : 0.f;
  red[t] = ea_; __syncthreads();
  for (int s = 64; s > 0; s >>= 1) { if (t < s) red[t] += red[t + s]; __syncthreads(); }
  const float asum = red[0]; __syncthreads();

  red[t] = p; __syncthreads();
  for (int s = 64; s > 0; s >>= 1) { if (t < s) red[t] = fmaxf(red[t], red[t + s]); __syncthreads(); }
  const float pmax = red[0]; __syncthreads();
  const float ep_ = (t < GPN) ? __expf(p - pmax) : 0.f;
  red[t] = ep_; __syncthreads();
  for (int s = 64; s > 0; s >>= 1) { if (t < s) red[t] += red[t + s]; __syncthreads(); }
  const float psum = red[0];

  if (t < GPN) w[e] = (ea_ / asum + wij[e] + ep_ / psum) * (1.f / 3.f);
}

// Pij_model: 3 -> 8 (ELU) -> 64, scalar (tiny)
__global__ void k_pijfeat(const float* __restrict__ Pij,
                          const float* __restrict__ W0, const float* __restrict__ b0,
                          const float* __restrict__ g0, const float* __restrict__ be0,
                          const float* __restrict__ W1, const float* __restrict__ b1,
                          const float* __restrict__ g1, const float* __restrict__ be1,
                          float* __restrict__ pf, int Nn) {
  int n = blockIdx.x * 256 + threadIdx.x;
  if (n >= Nn) return;
  const float p0 = Pij[n * 3], p1 = Pij[n * 3 + 1], p2 = Pij[n * 3 + 2];
  float h1[8];
#pragma unroll
  for (int o = 0; o < 8; ++o) {
    float s = p0 * W0[o * 3] + p1 * W0[o * 3 + 1] + p2 * W0[o * 3 + 2];
    s = (s + b0[o]) * (g0[o] * kBNInv) + be0[o];
    h1[o] = (s > 0.f) ? s : (__expf(s) - 1.f);
  }
  for (int f = 0; f < 64; ++f) {
    float s = 0.f;
#pragma unroll
    for (int o = 0; o < 8; ++o) s += h1[o] * W1[f * 8 + o];
    pf[(size_t)n * 64 + f] = (s + b1[f]) * (g1[f] * kBNInv) + be1[f];
  }
}

// u_out[b] = sum_{n in graph b} w[n] * pf[n] * x_new[n]  (one block/graph, 64 thr)
__global__ void k_uout(const float* __restrict__ w, const float* __restrict__ pf,
                       const half_t* __restrict__ xn, half_t* __restrict__ uout) {
  const int b = blockIdx.x, f = threadIdx.x;
  float acc = 0.f;
  for (int i = 0; i < GPN; ++i) {
    const int n = b * GPN + i;
    acc += w[n] * pf[(size_t)n * 64 + f] * (float)xn[(size_t)n * 64 + f];
  }
  uout[b * 64 + f] = (half_t)acc;
}

// collect per-layer [x_new[first node], u_new] into feat[B, 512]
__global__ void k_collect(const half_t* __restrict__ xn, const half_t* __restrict__ un,
                          float* __restrict__ feat, int layer) {
  const int b = blockIdx.x, f = threadIdx.x;
  feat[(size_t)b * 512 + layer * 64 + f] = (float)xn[(size_t)(b * GPN) * 64 + f];
  feat[(size_t)b * 512 + 256 + layer * 64 + f] = (float)un[b * 64 + f];
}

// classifier stage 1: h = ELU(BN(feat @ W1^T + b1))
__global__ void k_clf1(const float* __restrict__ feat, const float* __restrict__ W1,
                       const float* __restrict__ b1, const float* __restrict__ g1,
                       const float* __restrict__ be1, float* __restrict__ h) {
  int idx = blockIdx.x * 256 + threadIdx.x;
  if (idx >= NGRAPH * 64) return;
  const int b = idx >> 6, o = idx & 63;
  float s = 0.f;
  for (int c = 0; c < 512; ++c) s += feat[(size_t)b * 512 + c] * W1[(size_t)o * 512 + c];
  s = (s + b1[o]) * (g1[o] * kBNInv) + be1[o];
  h[idx] = (s > 0.f) ? s : (__expf(s) - 1.f);
}

// classifier stage 2: logits + softmax, take class 1
__global__ void k_clf2(const float* __restrict__ h, const float* __restrict__ W2,
                       const float* __restrict__ b2, float* __restrict__ out) {
  int b = blockIdx.x * 256 + threadIdx.x;
  if (b >= NGRAPH) return;
  float z0 = b2[0], z1 = b2[1];
  for (int c = 0; c < 64; ++c) {
    const float hv = h[b * 64 + c];
    z0 += hv * W2[c];
    z1 += hv * W2[64 + c];
  }
  const float m = fmaxf(z0, z1);
  const float e0 = __expf(z0 - m), e1 = __expf(z1 - m);
  out[b] = e1 / (e0 + e1);
}

// ---------------------------------------------------------------------------
// Host orchestration
// ---------------------------------------------------------------------------
extern "C" void kernel_launch(void* const* d_in, const int* in_sizes, int n_in,
                              void* d_out, int out_size, void* d_ws, size_t ws_size,
                              hipStream_t stream) {
  (void)in_sizes; (void)n_in; (void)out_size; (void)ws_size;
  const float* x     = (const float*)d_in[0];
  const float* dij   = (const float*)d_in[1];
  const int*   ei    = (const int*)d_in[2];
  const float* ea    = (const float*)d_in[3];
  const int*   gi    = (const int*)d_in[4];
  const float* wij   = (const float*)d_in[5];
  const float* Pij   = (const float*)d_in[6];
  const float* u     = (const float*)d_in[7];
  const int*   batch = (const int*)d_in[8];
  // d_in[9] = pos (unused by the network)
  const float* clfW1 = (const float*)d_in[10];
  const float* clfW2 = (const float*)d_in[11];
  const float* clfb1 = (const float*)d_in[12];
  const float* clfb2 = (const float*)d_in[13];
  const float* clfbe = (const float*)d_in[14];
  const float* clfg  = (const float*)d_in[15];
  // params['layers'][l] leaves in sorted-key flatten order; per-block: W,b,be,g
  auto LF = [&](int l, int o) -> const float* { return (const float*)d_in[16 + 62 * l + o]; };

  // workspace bump allocator
  char* wsb = (char*)d_ws;
  size_t off = 0;
  auto alloc = [&](size_t bytes) -> void* {
    void* p = (void*)(wsb + off);
    off += (bytes + 255) & ~(size_t)255;
    return p;
  };
  half_t* xhA  = (half_t*)alloc((size_t)NODES * 64 * 2);
  half_t* xhB  = (half_t*)alloc((size_t)NODES * 64 * 2);
  half_t* eaA  = (half_t*)alloc((size_t)EDGES * 64 * 2);
  half_t* eaB  = (half_t*)alloc((size_t)EDGES * 64 * 2);
  half_t* uhA  = (half_t*)alloc((size_t)NGRAPH * 64 * 2);
  half_t* uhB  = (half_t*)alloc((size_t)NGRAPH * 64 * 2);
  half_t* e_t1 = (half_t*)alloc((size_t)EDGES * 64 * 2);
  half_t* e_out= (half_t*)alloc((size_t)EDGES * 64 * 2);
  half_t* n_t1 = (half_t*)alloc((size_t)NODES * 64 * 2);
  half_t* x_out= (half_t*)alloc((size_t)NODES * 64 * 2);
  float*  agg  = (float*)alloc((size_t)NODES * 64 * 4);
  float*  attb = (float*)alloc((size_t)NODES * 4);
  float*  pwb  = (float*)alloc((size_t)NODES * 4);
  float*  wb   = (float*)alloc((size_t)NODES * 4);
  float*  pf   = (float*)alloc((size_t)NODES * 64 * 4);
  half_t* uout = (half_t*)alloc((size_t)NGRAPH * 64 * 2);
  half_t* g_t1 = (half_t*)alloc((size_t)NGRAPH * 64 * 2);
  half_t* g_t2 = (half_t*)alloc((size_t)NGRAPH * 64 * 2);
  half_t* g_o  = (half_t*)alloc((size_t)NGRAPH * 64 * 2);
  float*  feat = (float*)alloc((size_t)NGRAPH * 512 * 4);
  float*  hbuf = (float*)alloc((size_t)NGRAPH * 64 * 4);
  half_t* w16  = (half_t*)alloc((size_t)4 * 90112 * 2);

  // f16 weight conversion (leaf offsets: W of each block used by WMMA GEMMs)
  // j: 0 em0, 1 em1, 2 emlp0, 3 emlp1, 4 nm0, 5 nm1, 6 nmlp0, 7 nmlp1,
  //    8 gm0, 9 gm1, 10 gm2, 11 gmlp0, 12 gmlp1
  static const int wleaf[13] = {17, 21, 9, 13, 53, 57, 45, 49, 33, 37, 41, 25, 29};
  static const int welem[13] = {16384, 4096, 8192, 4096, 12288, 4096, 8192, 4096,
                                8192, 4096, 4096, 8192, 4096};
  half_t* WP[4][13];
  {
    size_t woff = 0;
    for (int l = 0; l < 4; ++l)
      for (int j = 0; j < 13; ++j) {
        WP[l][j] = w16 + woff;
        k_f2h<<<cdiv(welem[j], 256), 256, 0, stream>>>(LF(l, wleaf[j]), WP[l][j], welem[j]);
        woff += (size_t)welem[j];
      }
  }

  // initial activations -> f16
  k_f2h<<<cdiv(NODES * 64, 256), 256, 0, stream>>>(x, xhA, NODES * 64);
  k_f2h<<<cdiv(EDGES * 64, 256), 256, 0, stream>>>(ea, eaA, EDGES * 64);
  k_f2h<<<cdiv(NGRAPH * 64, 256), 256, 0, stream>>>(u, uhA, NGRAPH * 64);

  half_t* xh_c = xhA; half_t* xh_n = xhB;
  half_t* ea_c = eaA; half_t* ea_n = eaB;
  half_t* uh_c = uhA; half_t* uh_n = uhB;

  const int gE = cdiv(EDGES, 32);   // 5000
  const int gN = cdiv(NODES, 32);   // 625
  const int gB = cdiv(NGRAPH, 32);  // 7

  for (int l = 0; l < 4; ++l) {
    // ---- edge update ----
    k_edge_gemm<<<gE, 256, 0, stream>>>(xh_c, ea_c, uh_c, ei, batch, dij,
                                        WP[l][0], LF(l, 18), LF(l, 20), LF(l, 19),
                                        e_t1, EDGES);
    k_gemm<64, false><<<gE, 256, 0, stream>>>(e_t1, WP[l][1], LF(l, 22), LF(l, 24),
                                              LF(l, 23), e_out, EDGES);
    k_cat_gemm<true><<<gE, 256, 0, stream>>>(ea_c, e_out, WP[l][2], LF(l, 10),
                                             LF(l, 12), LF(l, 11), e_t1, EDGES);
    k_gemm<64, false><<<gE, 256, 0, stream>>>(e_t1, WP[l][3], LF(l, 14), LF(l, 16),
                                              LF(l, 15), ea_n, EDGES);
    // ---- node update ----
    k_zero<<<cdiv(NODES * 64, 256), 256, 0, stream>>>(agg, NODES * 64);
    k_edge_agg<<<cdiv(EDGES * 64, 256), 256, 0, stream>>>(ea_n, dij, ei, agg, EDGES);
    k_node_gemm<<<gN, 256, 0, stream>>>(xh_c, uh_c, agg, batch, WP[l][4],
                                        LF(l, 54), LF(l, 56), LF(l, 55), n_t1, NODES);
    k_gemm<64, false><<<gN, 256, 0, stream>>>(n_t1, WP[l][5], LF(l, 58), LF(l, 60),
                                              LF(l, 59), x_out, NODES);
    k_cat_gemm<true><<<gN, 256, 0, stream>>>(xh_c, x_out, WP[l][6], LF(l, 46),
                                             LF(l, 48), LF(l, 47), n_t1, NODES);
    k_gemm<64, false><<<gN, 256, 0, stream>>>(n_t1, WP[l][7], LF(l, 50), LF(l, 52),
                                              LF(l, 51), xh_n, NODES);
    // ---- global attention ----
    k_att_raw<<<cdiv(NODES, 256), 256, 0, stream>>>(xh_n, gi, LF(l, 8), LF(l, 61),
                                                    Pij, attb, pwb, NODES);
    k_softmax_w<<<NGRAPH, 128, 0, stream>>>(attb, pwb, wij, wb);
    k_pijfeat<<<cdiv(NODES, 256), 256, 0, stream>>>(Pij, LF(l, 0), LF(l, 1), LF(l, 3),
                                                    LF(l, 2), LF(l, 4), LF(l, 5),
                                                    LF(l, 7), LF(l, 6), pf, NODES);
    k_uout<<<NGRAPH, 64, 0, stream>>>(wb, pf, xh_n, uout);
    // ---- global update ----
    k_cat_gemm<true><<<gB, 256, 0, stream>>>(uh_c, uout, WP[l][8], LF(l, 34),
                                             LF(l, 36), LF(l, 35), g_t1, NGRAPH);
    k_gemm<64, true><<<gB, 256, 0, stream>>>(g_t1, WP[l][9], LF(l, 38), LF(l, 40),
                                             LF(l, 39), g_t2, NGRAPH);
    k_gemm<64, false><<<gB, 256, 0, stream>>>(g_t2, WP[l][10], LF(l, 42), LF(l, 44),
                                              LF(l, 43), g_o, NGRAPH);
    k_cat_gemm<true><<<gB, 256, 0, stream>>>(uh_c, g_o, WP[l][11], LF(l, 26),
                                             LF(l, 28), LF(l, 27), g_t1, NGRAPH);
    k_gemm<64, false><<<gB, 256, 0, stream>>>(g_t1, WP[l][12], LF(l, 30), LF(l, 32),
                                              LF(l, 31), uh_n, NGRAPH);
    // ---- collect per-layer features ----
    k_collect<<<NGRAPH, 64, 0, stream>>>(xh_n, uh_n, feat, l);
    // ping-pong
    half_t* t;
    t = xh_c; xh_c = xh_n; xh_n = t;
    t = ea_c; ea_c = ea_n; ea_n = t;
    t = uh_c; uh_c = uh_n; uh_n = t;
  }

  // classifier
  k_clf1<<<cdiv(NGRAPH * 64, 256), 256, 0, stream>>>(feat, clfW1, clfb1, clfg, clfbe, hbuf);
  k_clf2<<<cdiv(NGRAPH, 256), 256, 0, stream>>>(hbuf, clfW2, clfb2, (float*)d_out);
}
